// Attention_20298015441502
// MI455X (gfx1250) — matrix operand order repaired and checked by reading, not compiled, not run on hardware
//
#include <hip/hip_runtime.h>
#include <hip/hip_bf16.h>

// ---------- types ----------
typedef __attribute__((ext_vector_type(16))) __bf16 v16bf;
typedef __attribute__((ext_vector_type(8)))  float  v8f;

union FragB16 {            // 8 VGPRs = 16 bf16 elements
    v16bf v;
    unsigned int u[8];
    unsigned short s[16];
};

static __device__ __forceinline__ v8f wmma_bf16(v16bf a, v16bf b, v8f c) {
    return __builtin_amdgcn_wmma_f32_16x16x32_bf16(
        false, a, false, b, (short)0, c, false, false);
}

static __device__ __forceinline__ unsigned short f2bf(float f) {
    unsigned int x = __float_as_uint(f);
    x += 0x7FFFu + ((x >> 16) & 1u);   // round-to-nearest-even
    return (unsigned short)(x >> 16);
}

// CDNA5 async DMA: global -> LDS, 16 B per lane, tracked by ASYNCcnt.
// (inline asm: portable across ROCm7.2 / amdgpu-toolchain builtin arity skew)
static __device__ __forceinline__ void async_ld_b128(void* lds, const void* g) {
    unsigned loff = (unsigned)(unsigned long long)lds;           // LDS offset
    unsigned long long ga = (unsigned long long)g;
    asm volatile("global_load_async_to_lds_b128 %0, %1, off"
                 :: "v"(loff), "v"(ga) : "memory");
}
static __device__ __forceinline__ void wait_async0() {
    asm volatile("s_wait_asynccnt 0x0" ::: "memory");
}

// A-matrix (16x32 bf16) per-lane K index for VGPR v (pairs), ISA 7.12.2
static __device__ __forceinline__ int kA(int v, int hi) {
    return (v < 4 ? 2 * v : 16 + 2 * (v - 4)) + (hi ? 8 : 0);
}
// B-matrix (32x16 bf16) per-lane K index for VGPR v (pairs)
static __device__ __forceinline__ int kB(int v, int hi) {
    return (v < 4 ? 2 * v : 16 + 2 * (v - 4)) + (hi ? 8 : 0);
}

// ---------- constants ----------
#define SEQ    4096
#define CH     1024
#define HEADS  16
#define HD     64
#define WIN    256
#define ROWS   8192          // B*S
#define QKVC   3072

// ---------- kernel 1: fp32 -> raw bf16 ----------
__global__ void k_cvt_bf16(const float* __restrict__ src,
                           unsigned short* __restrict__ dst, int n) {
    int i = blockIdx.x * blockDim.x + threadIdx.x;
    if (i < n) dst[i] = f2bf(src[i]);
}

// ---------- kernel 2: RMSNorm (one row per block) ----------
__global__ void k_rmsnorm(const float* __restrict__ x,
                          const float* __restrict__ scale,
                          unsigned short* __restrict__ h) {
    __shared__ float red[256];
    int row = blockIdx.x;
    int tid = threadIdx.x;
    const float* xr = x + (size_t)row * CH;
    float s = 0.f;
    for (int i = tid; i < CH; i += 256) { float v = xr[i]; s += v * v; }
    red[tid] = s;
    __syncthreads();
    for (int off = 128; off > 0; off >>= 1) {
        if (tid < off) red[tid] += red[tid + off];
        __syncthreads();
    }
    float inv = rsqrtf(red[0] / (float)CH + 1e-5f);
    unsigned short* hr = h + (size_t)row * CH;
    for (int i = tid; i < CH; i += 256)
        hr[i] = f2bf(xr[i] * scale[i] * inv);
}

// ---------- kernel 3/5: bf16 WMMA GEMM, C = A @ B^T ----------
// Double-buffered LDS, async global->LDS DMA pipeline (no VGPR staging), BK=64.
// A: [ROWS][K] bf16 raw, Bw: [N][K] bf16 raw (row n = output column n)
// mode 0: RoPE(q,k cols) epilogue -> bf16 out; mode 1: +residual -> fp32 out
#define BM 128
#define BN 128
#define BK 64
__global__ void __launch_bounds__(256)
k_gemm_bf16(const unsigned short* __restrict__ A, int lda,
            const unsigned short* __restrict__ Bw, int ldb,
            int N, int K, int mode,
            const float* __restrict__ resid,
            float* __restrict__ outF,
            unsigned short* __restrict__ outBF, int ldo) {
    __shared__ unsigned short ldsA[2][BM][BK];   // 32 KB
    __shared__ unsigned short ldsB[2][BN][BK];   // 32 KB

    const int tid  = threadIdx.x;
    const int wid  = tid >> 5;
    const int lane = tid & 31;
    const int l16  = lane & 15;
    const int hi   = lane >> 4;

    const int m0 = blockIdx.y * BM;
    const int n0 = blockIdx.x * BN;
    const int wm = (wid & 3) * 32;    // wave row offset (32 rows)
    const int wn = (wid >> 2) * 64;   // wave col offset (64 cols, head aligned)

    // staging assignment: each thread owns one row-half (64 bytes) of each tile
    const int srow  = tid >> 1;          // 0..127
    const int shalf = (tid & 1) * 32;    // 0 or 32 (ushort offset)
    const unsigned short* aSrc = A  + (size_t)(m0 + srow) * lda + shalf;
    const unsigned short* bSrc = Bw + (size_t)(n0 + srow) * ldb + shalf;

    v8f acc[2][4];
    v8f z = {};
    for (int a = 0; a < 2; ++a)
        for (int b = 0; b < 4; ++b) acc[a][b] = z;

    const int nk = K / BK;

    // prologue: async-stage tile 0 into buffer 0
    for (int j = 0; j < 4; ++j) {
        async_ld_b128(&ldsA[0][srow][shalf + j * 8], aSrc + j * 8);
        async_ld_b128(&ldsB[0][srow][shalf + j * 8], bSrc + j * 8);
    }
    wait_async0();
    __syncthreads();

    for (int kt = 0; kt < nk; ++kt) {
        const int buf = kt & 1;

        // launch async DMA for next tile into the other buffer
        if (kt + 1 < nk) {
            const unsigned short* an = aSrc + (kt + 1) * BK;
            const unsigned short* bn = bSrc + (kt + 1) * BK;
            for (int j = 0; j < 4; ++j) {
                async_ld_b128(&ldsA[buf ^ 1][srow][shalf + j * 8], an + j * 8);
                async_ld_b128(&ldsB[buf ^ 1][srow][shalf + j * 8], bn + j * 8);
            }
        }
        // prefetch tile kt+2 into L2 (global_prefetch_b8)
        if (kt + 2 < nk) {
            __builtin_prefetch(aSrc + (kt + 2) * BK, 0, 3);
            __builtin_prefetch(bSrc + (kt + 2) * BK, 0, 3);
        }

        // compute on current buffer: 2 K-chunks of 32 -> 16 WMMAs
        for (int kc = 0; kc < 2; ++kc) {
            FragB16 af[2], bf[4];
            for (int ms = 0; ms < 2; ++ms) {
                int r = wm + ms * 16 + l16;
                for (int v = 0; v < 8; ++v)
                    af[ms].u[v] =
                        *(const unsigned int*)&ldsA[buf][r][kc * 32 + kA(v, hi)];
            }
            for (int ns = 0; ns < 4; ++ns) {
                int c = wn + ns * 16 + l16;
                for (int v = 0; v < 8; ++v)
                    bf[ns].u[v] =
                        *(const unsigned int*)&ldsB[buf][c][kc * 32 + kB(v, hi)];
            }
            for (int ms = 0; ms < 2; ++ms)
                for (int ns = 0; ns < 4; ++ns)
                    acc[ms][ns] = wmma_bf16(af[ms].v, bf[ns].v, acc[ms][ns]);
        }

        // publish next buffer
        wait_async0();
        __syncthreads();
    }

    // ---- epilogue ----
    const int colBase = n0 + wn;                 // head-aligned 64-wide span
    const int rowBase = m0 + wm;

    if (mode == 0) {
        int region = colBase / CH;               // 0=q 1=k 2=v
        if (region < 2) {
            int head = (colBase % CH) / HD;
            for (int ms = 0; ms < 2; ++ms) {
                for (int nsl = 0; nsl < 2; ++nsl) {
                    int j = nsl * 16 + l16;      // rotary pair index (d < 32)
                    float invf = __expf(-(float)(2 * j) * (1.0f / 64.0f) *
                                        9.210340371976184f);
                    float ang = (float)head * invf;
                    float c = __cosf(ang), s = __sinf(ang);
                    for (int v = 0; v < 8; ++v) {
                        float a0 = acc[ms][nsl][v];
                        float a2 = acc[ms][nsl + 2][v];
                        float o0 = a0 * c - a2 * s;
                        float o2 = a2 * c + a0 * s;
                        int row = rowBase + ms * 16 + v + (hi ? 8 : 0);
                        outBF[(size_t)row * ldo + colBase + nsl * 16 + l16]       = f2bf(o0);
                        outBF[(size_t)row * ldo + colBase + (nsl + 2) * 16 + l16] = f2bf(o2);
                    }
                }
            }
        } else {
            for (int ms = 0; ms < 2; ++ms)
                for (int ns = 0; ns < 4; ++ns)
                    for (int v = 0; v < 8; ++v) {
                        int row = rowBase + ms * 16 + v + (hi ? 8 : 0);
                        outBF[(size_t)row * ldo + colBase + ns * 16 + l16] =
                            f2bf(acc[ms][ns][v]);
                    }
        }
    } else {
        for (int ms = 0; ms < 2; ++ms)
            for (int ns = 0; ns < 4; ++ns)
                for (int v = 0; v < 8; ++v) {
                    int row = rowBase + ms * 16 + v + (hi ? 8 : 0);
                    size_t idx = (size_t)row * ldo + colBase + ns * 16 + l16;
                    outF[idx] = acc[ms][ns][v] + resid[idx];
                }
    }
}

// ---------- kernel 4: banded flash attention ----------
// one WG per (batch, head, 256-query block); 8 waves x 32 queries
__global__ void __launch_bounds__(256)
k_attn(const unsigned short* __restrict__ qkv,
       unsigned short* __restrict__ ao) {
    __shared__ unsigned short ldsK[64][64];     // [key][d]
    __shared__ unsigned short ldsVt[64][64];    // [d][key] (transposed)
    __shared__ unsigned short ldsP[8][16 * 64]; // per-wave P tile

    const int NB   = SEQ / WIN;                 // 16
    const int id   = blockIdx.x;
    const int b    = id / (HEADS * NB);
    const int h    = (id / NB) % HEADS;
    const int blk  = id % NB;

    const int tid  = threadIdx.x;
    const int wid  = tid >> 5;
    const int lane = tid & 31;
    const int l16  = lane & 15;
    const int hi   = lane >> 4;

    const int rowQ0 = b * SEQ + blk * WIN;
    const float scale = 0.125f;                 // 1/sqrt(64)

    // staging slot for cooperative K/V chunk loads
    const int kr = tid >> 2, sg = tid & 3;
    const unsigned short* kBase =
        qkv + (size_t)(b * SEQ) * QKVC + CH + h * HD + sg * 16;       // + kg*QKVC
    const unsigned short* vBase =
        qkv + (size_t)(b * SEQ) * QKVC + 2 * CH + h * HD + sg * 16;   // + kg*QKVC

    // Q fragments (A layout), direct from global
    FragB16 qf[2][2];
    for (int ms = 0; ms < 2; ++ms) {
        int qi = wid * 32 + ms * 16 + l16;
        const unsigned short* qr = qkv + (size_t)(rowQ0 + qi) * QKVC + h * HD;
        for (int kc = 0; kc < 2; ++kc)
            for (int v = 0; v < 8; ++v)
                qf[ms][kc].u[v] = *(const unsigned int*)&qr[kc * 32 + kA(v, hi)];
    }

    float mrow[2][8], lrow[2][8];
    v8f Oacc[2][4];
    v8f z = {};
    for (int ms = 0; ms < 2; ++ms)
        for (int v = 0; v < 8; ++v) { mrow[ms][v] = -3e38f; lrow[ms][v] = 0.f; }
    for (int ms = 0; ms < 2; ++ms)
        for (int dt = 0; dt < 4; ++dt) Oacc[ms][dt] = z;

    unsigned short* Pw = &ldsP[wid][0];

    for (int kc64 = 0; kc64 < 12; ++kc64) {
        // chunk validity is uniform: chunk base is a multiple of 64
        const int kgBase = blk * WIN - WIN + kc64 * 64;
        const bool chunkValid = (kgBase >= 0) && (kgBase < SEQ);
        const int kg = kgBase + kr;

        __syncthreads();   // previous iteration's reads complete
        if (chunkValid) {
            // K chunk: async DMA straight into LDS ([key][d] row-major copy)
            const unsigned short* ks = kBase + (size_t)kg * QKVC;
            async_ld_b128(&ldsK[kr][sg * 16],     ks);
            async_ld_b128(&ldsK[kr][sg * 16 + 8], ks + 8);
            // V chunk: register path (needs transpose into [d][key])
            const uint4* vs = (const uint4*)(vBase + (size_t)kg * QKVC);
            uint4 vp0 = vs[0], vp1 = vs[1];
            unsigned short tmp[16];
            *(uint4*)&tmp[0] = vp0;
            *(uint4*)&tmp[8] = vp1;
            for (int e = 0; e < 16; ++e) ldsVt[sg * 16 + e][kr] = tmp[e];
            // prefetch next chunk
            if (kc64 + 1 < 12 && kgBase + 64 < SEQ) {
                __builtin_prefetch(kBase + (size_t)(kg + 64) * QKVC, 0, 3);
                __builtin_prefetch(vBase + (size_t)(kg + 64) * QKVC, 0, 3);
            }
        }
        wait_async0();
        __syncthreads();
        // For invalid chunks LDS holds stale data: scores are masked to -3e38,
        // and any O/l garbage accumulated while m==-3e38 is wiped by the
        // exp(m_old - m_new) = 0 rescale at the first valid chunk.

        for (int ms = 0; ms < 2; ++ms) {
            // S = Q @ K^T for this 16x64 stripe
            v8f sacc[4];
            for (int nt = 0; nt < 4; ++nt) sacc[nt] = z;
            for (int nt = 0; nt < 4; ++nt) {
                for (int kc2 = 0; kc2 < 2; ++kc2) {
                    FragB16 kb;
                    int kcol = nt * 16 + l16;
                    for (int v = 0; v < 8; ++v)
                        kb.u[v] = *(const unsigned int*)&ldsK[kcol][kc2 * 32 + kB(v, hi)];
                    sacc[nt] = wmma_bf16(qf[ms][kc2].v, kb.v, sacc[nt]);
                }
            }
            // scale + band mask
            for (int nt = 0; nt < 4; ++nt)
                for (int v = 0; v < 8; ++v) {
                    int qi = wid * 32 + ms * 16 + v + (hi ? 8 : 0);
                    int j  = kc64 * 64 + nt * 16 + l16;
                    int kgj = blk * WIN - WIN + j;
                    bool ok = (j >= qi) && (j <= qi + 2 * WIN) && (kgj >= 0) && (kgj < SEQ);
                    float sv = sacc[nt][v] * scale;
                    sacc[nt][v] = ok ? sv : -3e38f;
                }
            // online softmax: row max over 4 tiles + 16 lanes
            float mnew[8], rs[8], lsum[8];
            for (int v = 0; v < 8; ++v) {
                float t = sacc[0][v];
                t = fmaxf(t, sacc[1][v]);
                t = fmaxf(t, sacc[2][v]);
                t = fmaxf(t, sacc[3][v]);
                for (int msk = 1; msk < 16; msk <<= 1)
                    t = fmaxf(t, __shfl_xor(t, msk, 16));
                mnew[v] = fmaxf(mrow[ms][v], t);
                rs[v]   = __expf(mrow[ms][v] - mnew[v]);
                mrow[ms][v] = mnew[v];
                lsum[v] = 0.f;
            }
            for (int nt = 0; nt < 4; ++nt)
                for (int v = 0; v < 8; ++v) {
                    float p = __expf(sacc[nt][v] - mnew[v]);
                    sacc[nt][v] = p;
                    lsum[v] += p;
                }
            for (int v = 0; v < 8; ++v) {
                float t = lsum[v];
                for (int msk = 1; msk < 16; msk <<= 1)
                    t += __shfl_xor(t, msk, 16);
                lrow[ms][v] = lrow[ms][v] * rs[v] + t;
            }
            // P -> per-wave LDS (C layout -> A layout round trip)
            for (int nt = 0; nt < 4; ++nt)
                for (int v = 0; v < 8; ++v) {
                    int r = v + (hi ? 8 : 0);
                    Pw[r * 64 + nt * 16 + l16] = f2bf(sacc[nt][v]);
                }
            // rescale O, then O += P @ V
            for (int dt = 0; dt < 4; ++dt)
                for (int v = 0; v < 8; ++v)
                    Oacc[ms][dt][v] *= rs[v];
            for (int dt = 0; dt < 4; ++dt) {
                for (int kc2 = 0; kc2 < 2; ++kc2) {
                    FragB16 pa, vb;
                    for (int v = 0; v < 8; ++v)
                        pa.u[v] = *(const unsigned int*)&Pw[l16 * 64 + kc2 * 32 + kA(v, hi)];
                    int dcol = dt * 16 + l16;
                    for (int v = 0; v < 8; ++v)
                        vb.u[v] = *(const unsigned int*)&ldsVt[dcol][kc2 * 32 + kB(v, hi)];
                    Oacc[ms][dt] = wmma_bf16(pa.v, vb.v, Oacc[ms][dt]);
                }
            }
        }
    }

    // normalize + store (B,S,H*D) bf16
    for (int ms = 0; ms < 2; ++ms)
        for (int dt = 0; dt < 4; ++dt)
            for (int v = 0; v < 8; ++v) {
                int qi  = wid * 32 + ms * 16 + v + (hi ? 8 : 0);
                int row = rowQ0 + qi;
                int col = h * HD + dt * 16 + l16;
                float o = Oacc[ms][dt][v] / lrow[ms][v];
                ao[(size_t)row * CH + col] = f2bf(o);
            }
}

// ---------- host ----------
extern "C" void kernel_launch(void* const* d_in, const int* in_sizes, int n_in,
                              void* d_out, int out_size, void* d_ws, size_t ws_size,
                              hipStream_t stream) {
    const float* x      = (const float*)d_in[0];
    const float* w_qkv  = (const float*)d_in[2];
    const float* w_out  = (const float*)d_in[3];
    const float* nscale = (const float*)d_in[4];
    float* out = (float*)d_out;

    unsigned char* ws = (unsigned char*)d_ws;
    size_t off = 0;
    unsigned short* h_bf    = (unsigned short*)(ws + off); off += (size_t)ROWS * CH * 2;
    unsigned short* wqkv_bf = (unsigned short*)(ws + off); off += (size_t)QKVC * CH * 2;
    unsigned short* wout_bf = (unsigned short*)(ws + off); off += (size_t)CH * CH * 2;
    unsigned short* qkv_bf  = (unsigned short*)(ws + off); off += (size_t)ROWS * QKVC * 2;
    unsigned short* ao_bf   = (unsigned short*)(ws + off); off += (size_t)ROWS * CH * 2;

    {
        int n = QKVC * CH;
        k_cvt_bf16<<<(n + 255) / 256, 256, 0, stream>>>(w_qkv, wqkv_bf, n);
        n = CH * CH;
        k_cvt_bf16<<<(n + 255) / 256, 256, 0, stream>>>(w_out, wout_bf, n);
    }
    k_rmsnorm<<<ROWS, 256, 0, stream>>>(x, nscale, h_bf);

    // QKV projection with fused head-indexed RoPE
    k_gemm_bf16<<<dim3(QKVC / BN, ROWS / BM), 256, 0, stream>>>(
        h_bf, CH, wqkv_bf, CH, QKVC, CH, /*mode=*/0,
        nullptr, nullptr, qkv_bf, QKVC);

    // banded attention: 2 * 16 * 16 = 512 workgroups
    k_attn<<<2 * HEADS * (SEQ / WIN), 256, 0, stream>>>(qkv_bf, ao_bf);

    // output projection + residual
    k_gemm_bf16<<<dim3(CH / BN, ROWS / BM), 256, 0, stream>>>(
        ao_bf, CH, wout_bf, CH, CH, CH, /*mode=*/1,
        x, out, nullptr, CH);
}
